// SelfAttentionModule_65481071405046
// MI455X (gfx1250) — matrix operand order repaired and checked by reading, not compiled, not run on hardware
//
#include <hip/hip_runtime.h>

// ---------------------------------------------------------------------------
// Types for CDNA5 WMMA (gfx1250, wave32)
// ---------------------------------------------------------------------------
typedef __bf16 bf16_t;
typedef __attribute__((ext_vector_type(16))) __bf16 v16bf;
typedef __attribute__((ext_vector_type(8)))  __bf16 v8bf;
typedef __attribute__((ext_vector_type(8)))  float  v8f;

#define BATCH    2
#define SEQ      4096
#define DMODEL   512
#define NHEADS   8
#define HDIM     64
#define ROWS     (BATCH * SEQ)          // 8192
#define QKV_N    (3 * DMODEL)           // 1536
#define BH       (BATCH * NHEADS)       // 16
#define SM_SCALE 0.125f                 // 1/sqrt(64)
#define LN_EPS   1e-5f

__device__ __forceinline__ v8f zero_v8f() {
  v8f z;
#pragma unroll
  for (int i = 0; i < 8; ++i) z[i] = 0.0f;
  return z;
}

__device__ __forceinline__ v16bf combine16(v8bf lo, v8bf hi) {
  v16bf r;
#pragma unroll
  for (int i = 0; i < 8; ++i) { r[i] = lo[i]; r[i + 8] = hi[i]; }
  return r;
}

// A-matrix fragment, 16x32 bf16, row-major source with leading dim `ld`.
// Layout: lanes 0-15 -> row=lane, K 0..15; lanes 16-31 -> row=lane-16,
// K 16..31.  Two contiguous 16-byte loads per lane.
__device__ __forceinline__ v16bf load_a_frag(const bf16_t* base_row0, int k0, int ld) {
  const int lane = threadIdx.x & 31;
  const int r    = lane & 15;
  const int kb   = (lane >> 4) << 4;      // 0 or 16
  const bf16_t* p = base_row0 + (size_t)r * ld + k0 + kb;
  v8bf lo = *reinterpret_cast<const v8bf*>(p);
  v8bf hi = *reinterpret_cast<const v8bf*>(p + 8);
  return combine16(lo, hi);
}

// B-matrix fragment, 32x16 bf16. Source stored K-major per column:
// element (k, col) at base[col*ld + k].  Layout used here (contiguous k, same as A): N = lane%16; lanes 0-15
// carry K 0..15, lanes 16-31 carry K 16..31 -> one contiguous 32-byte load.
__device__ __forceinline__ v16bf load_b_frag(const bf16_t* base, int col0, int k0, int ld) {
  const int lane = threadIdx.x & 31;
  const int c    = col0 + (lane & 15);
  const int kb   = (lane >> 4) << 4;      // 0 or 16
  return *reinterpret_cast<const v16bf*>(base + (size_t)c * ld + k0 + kb);
}

__device__ __forceinline__ v8f wmma_bf16(v16bf a, v16bf b, v8f c) {
  return __builtin_amdgcn_wmma_f32_16x16x32_bf16(
      /*neg_a=*/false, a, /*neg_b=*/false, b,
      /*c_mod=*/(short)0, c, /*reuse_a=*/false, /*reuse_b=*/false);
}

// ---------------------------------------------------------------------------
// Prep kernels: f32 -> bf16 convert, and weight transpose (to K-major columns)
// ---------------------------------------------------------------------------
__global__ void sa_convert_bf16(const float* __restrict__ src,
                                bf16_t* __restrict__ dst, int n) {
  int i = blockIdx.x * blockDim.x + threadIdx.x;
  if (i < n) dst[i] = (bf16_t)src[i];
}

// W[K,N] (row-major) -> Wt[N,K] bf16
__global__ void sa_transpose_bf16(const float* __restrict__ W,
                                  bf16_t* __restrict__ Wt, int K, int N) {
  int i = blockIdx.x * blockDim.x + threadIdx.x;
  if (i < K * N) {
    int k = i / N, n = i % N;
    Wt[(size_t)n * K + k] = (bf16_t)W[i];
  }
}

// ---------------------------------------------------------------------------
// QKV projection GEMM:  [8192 x 512] @ [512 x 1536] -> Q,K row-major, V transposed
// One wave computes a 16x64 tile. 4 waves / block.
// ---------------------------------------------------------------------------
__global__ void sa_qkv_gemm(const bf16_t* __restrict__ xb,
                            const bf16_t* __restrict__ WqkvT,
                            bf16_t* __restrict__ Q,
                            bf16_t* __restrict__ K,
                            bf16_t* __restrict__ Vt) {
  const int lane = threadIdx.x & 31;
  const int wid  = blockIdx.x * (blockDim.x >> 5) + (threadIdx.x >> 5);
  const int NT   = QKV_N / 64;              // 24
  const int m0   = (wid / NT) * 16;
  const int n0   = (wid % NT) * 64;

  v8f acc[4];
#pragma unroll
  for (int j = 0; j < 4; ++j) acc[j] = zero_v8f();

  const bf16_t* arow = xb + (size_t)m0 * DMODEL;
  for (int k0 = 0; k0 < DMODEL; k0 += 32) {
    v16bf a = load_a_frag(arow, k0, DMODEL);
#pragma unroll
    for (int j = 0; j < 4; ++j) {
      v16bf b = load_b_frag(WqkvT, n0 + j * 16, k0, DMODEL);
      acc[j] = wmma_bf16(a, b, acc[j]);
    }
  }

  // Epilogue: scatter into Q/K [BH, S, 64] or V transposed [BH, 64, S]
  const int hi  = lane >> 4;
  const int col = lane & 15;
  const int sec = n0 >> 9;                  // 0=q, 1=k, 2=v
  const int h   = (n0 & 511) >> 6;
#pragma unroll
  for (int j = 0; j < 4; ++j) {
#pragma unroll
    for (int r = 0; r < 8; ++r) {
      const int gm = m0 + r + 8 * hi;
      const int b  = gm >> 12;              // / SEQ
      const int s  = gm & (SEQ - 1);
      const int d  = j * 16 + col;
      const bf16_t v = (bf16_t)acc[j][r];
      const size_t bhb = (size_t)(b * NHEADS + h);
      if (sec == 0)      Q[(bhb * SEQ + s) * HDIM + d] = v;
      else if (sec == 1) K[(bhb * SEQ + s) * HDIM + d] = v;
      else               Vt[(bhb * HDIM + d) * SEQ + s] = v;
    }
  }
}

// ---------------------------------------------------------------------------
// Flash attention: one wave owns 16 query rows, streams 4096 keys in blocks of
// 32 with online softmax. Context written bf16 [B,S,D].
// ---------------------------------------------------------------------------
__global__ void sa_flash_attn(const bf16_t* __restrict__ Q,
                              const bf16_t* __restrict__ K,
                              const bf16_t* __restrict__ Vt,
                              bf16_t* __restrict__ ctx) {
  __shared__ __align__(32) bf16_t plds[4][16 * 32];   // 1 KB per wave

  const int lane  = threadIdx.x & 31;
  const int wslot = threadIdx.x >> 5;
  const int wid   = blockIdx.x * 4 + wslot;
  const int qt    = wid & 255;              // SEQ/16 tiles per (b,h)
  const int bh    = wid >> 8;               // 0..15
  const int q0    = qt * 16;

  const bf16_t* Qb = Q  + (size_t)bh * SEQ * HDIM;
  const bf16_t* Kb = K  + (size_t)bh * SEQ * HDIM;
  const bf16_t* Vb = Vt + (size_t)bh * HDIM * SEQ;
  bf16_t* plane = &plds[wslot][0];

  const int hi  = lane >> 4;
  const int col = lane & 15;

  v16bf qa0 = load_a_frag(Qb + (size_t)q0 * HDIM, 0,  HDIM);
  v16bf qa1 = load_a_frag(Qb + (size_t)q0 * HDIM, 32, HDIM);

  v8f acc[4];
#pragma unroll
  for (int t = 0; t < 4; ++t) acc[t] = zero_v8f();
  float mrow[8], lrow[8];
#pragma unroll
  for (int r = 0; r < 8; ++r) { mrow[r] = -1e30f; lrow[r] = 0.0f; }

  for (int kk = 0; kk < SEQ; kk += 32) {
#pragma unroll
    for (int sub = 0; sub < 2; ++sub) {
      const int key0 = kk + sub * 16;
      v16bf kb0 = load_b_frag(Kb, key0, 0,  HDIM);   // col = key, K-dim = d
      v16bf kb1 = load_b_frag(Kb, key0, 32, HDIM);
      v8f s = zero_v8f();
      s = wmma_bf16(qa0, kb0, s);
      s = wmma_bf16(qa1, kb1, s);
#pragma unroll
      for (int r = 0; r < 8; ++r) {
        float sv = s[r] * SM_SCALE;
        // row-wise max across the 16 lanes of this half-wave
        float bmax = sv;
        bmax = fmaxf(bmax, __shfl_xor(bmax, 1, 32));
        bmax = fmaxf(bmax, __shfl_xor(bmax, 2, 32));
        bmax = fmaxf(bmax, __shfl_xor(bmax, 4, 32));
        bmax = fmaxf(bmax, __shfl_xor(bmax, 8, 32));
        const float newm  = fmaxf(mrow[r], bmax);
        const float alpha = __expf(mrow[r] - newm);
        const float p     = __expf(sv - newm);
        float psum = p;
        psum += __shfl_xor(psum, 1, 32);
        psum += __shfl_xor(psum, 2, 32);
        psum += __shfl_xor(psum, 4, 32);
        psum += __shfl_xor(psum, 8, 32);
        lrow[r] = lrow[r] * alpha + psum;
        mrow[r] = newm;
        acc[0][r] *= alpha; acc[1][r] *= alpha;
        acc[2][r] *= alpha; acc[3][r] *= alpha;
        // stash P (C-layout element) into LDS for transpose to A-layout
        plane[(r + 8 * hi) * 32 + sub * 16 + col] = (bf16_t)p;
      }
    }
    // P tile 16x32 bf16 now in LDS; read back in A-fragment layout
    v16bf pa = load_a_frag(plane, 0, 32);
#pragma unroll
    for (int t = 0; t < 4; ++t) {
      v16bf vb = load_b_frag(Vb, t * 16, kk, SEQ);   // col = d, K-dim = key
      acc[t] = wmma_bf16(pa, vb, acc[t]);
    }
  }

  // Epilogue: normalize and write context [B,S,D]
  const int b = bh >> 3, h = bh & 7;
#pragma unroll
  for (int t = 0; t < 4; ++t) {
#pragma unroll
    for (int r = 0; r < 8; ++r) {
      const int s = q0 + r + 8 * hi;
      const float val = acc[t][r] / lrow[r];
      ctx[((size_t)(b * SEQ + s)) * DMODEL + h * HDIM + t * 16 + col] = (bf16_t)val;
    }
  }
}

// ---------------------------------------------------------------------------
// Out-projection GEMM + bias + residual -> f32 scratch
// ---------------------------------------------------------------------------
__global__ void sa_out_gemm(const bf16_t* __restrict__ ctx,
                            const bf16_t* __restrict__ WoutT,
                            const float* __restrict__ bout,
                            const float* __restrict__ x,
                            float* __restrict__ yres) {
  const int lane = threadIdx.x & 31;
  const int wid  = blockIdx.x * (blockDim.x >> 5) + (threadIdx.x >> 5);
  const int NT   = DMODEL / 64;             // 8
  const int m0   = (wid / NT) * 16;
  const int n0   = (wid % NT) * 64;

  v8f acc[4];
#pragma unroll
  for (int j = 0; j < 4; ++j) acc[j] = zero_v8f();

  const bf16_t* arow = ctx + (size_t)m0 * DMODEL;
  for (int k0 = 0; k0 < DMODEL; k0 += 32) {
    v16bf a = load_a_frag(arow, k0, DMODEL);
#pragma unroll
    for (int j = 0; j < 4; ++j) {
      v16bf b = load_b_frag(WoutT, n0 + j * 16, k0, DMODEL);
      acc[j] = wmma_bf16(a, b, acc[j]);
    }
  }

  const int hi  = lane >> 4;
  const int col = lane & 15;
#pragma unroll
  for (int j = 0; j < 4; ++j) {
#pragma unroll
    for (int r = 0; r < 8; ++r) {
      const int gm = m0 + r + 8 * hi;
      const int n  = n0 + j * 16 + col;
      const size_t idx = (size_t)gm * DMODEL + n;
      yres[idx] = acc[j][r] + bout[n] + x[idx];
    }
  }
}

// ---------------------------------------------------------------------------
// LayerNorm over D=512: one wave per row (16 elems/lane), wave-wide reductions
// ---------------------------------------------------------------------------
__global__ void sa_layernorm(const float* __restrict__ yres,
                             const float* __restrict__ gamma,
                             const float* __restrict__ beta,
                             float* __restrict__ out) {
  const int lane = threadIdx.x & 31;
  const int row  = blockIdx.x * (blockDim.x >> 5) + (threadIdx.x >> 5);
  const float* yr = yres + (size_t)row * DMODEL;

  float v[16];
  float sum = 0.0f;
#pragma unroll
  for (int i = 0; i < 16; ++i) { v[i] = yr[lane + i * 32]; sum += v[i]; }
#pragma unroll
  for (int m = 1; m < 32; m <<= 1) sum += __shfl_xor(sum, m, 32);
  const float mean = sum * (1.0f / DMODEL);

  float var = 0.0f;
#pragma unroll
  for (int i = 0; i < 16; ++i) { float d = v[i] - mean; var += d * d; }
#pragma unroll
  for (int m = 1; m < 32; m <<= 1) var += __shfl_xor(var, m, 32);
  const float rstd = rsqrtf(var * (1.0f / DMODEL) + LN_EPS);

  float* orow = out + (size_t)row * DMODEL;
#pragma unroll
  for (int i = 0; i < 16; ++i) {
    const int d = lane + i * 32;
    orow[d] = (v[i] - mean) * rstd * gamma[d] + beta[d];
  }
}

// ---------------------------------------------------------------------------
// Host launch
// ---------------------------------------------------------------------------
extern "C" void kernel_launch(void* const* d_in, const int* in_sizes, int n_in,
                              void* d_out, int out_size, void* d_ws, size_t ws_size,
                              hipStream_t stream) {
  const float* x     = (const float*)d_in[0];   // [2,4096,512]
  const float* Wqkv  = (const float*)d_in[1];   // [512,1536]
  const float* Wout  = (const float*)d_in[2];   // [512,512]
  const float* bout  = (const float*)d_in[3];   // [512]
  const float* gamma = (const float*)d_in[4];   // [512]
  const float* beta  = (const float*)d_in[5];   // [512]
  float* out = (float*)d_out;

  char* ws = (char*)d_ws;
  size_t off = 0;
  auto carve = [&](size_t bytes) -> void* {
    void* p = ws + off;
    off = (off + bytes + 255) & ~(size_t)255;
    return p;
  };
  bf16_t* xb     = (bf16_t*)carve((size_t)ROWS * DMODEL * 2);      // 8 MB
  bf16_t* WqkvT  = (bf16_t*)carve((size_t)QKV_N * DMODEL * 2);     // 1.5 MB
  bf16_t* WoutT  = (bf16_t*)carve((size_t)DMODEL * DMODEL * 2);    // 0.5 MB
  bf16_t* Qb     = (bf16_t*)carve((size_t)BH * SEQ * HDIM * 2);    // 8 MB
  bf16_t* Kb     = (bf16_t*)carve((size_t)BH * SEQ * HDIM * 2);    // 8 MB
  bf16_t* Vtb    = (bf16_t*)carve((size_t)BH * HDIM * SEQ * 2);    // 8 MB
  bf16_t* ctx    = (bf16_t*)carve((size_t)ROWS * DMODEL * 2);      // 8 MB
  float*  yres   = (float*)carve((size_t)ROWS * DMODEL * 4);       // 16 MB

  const int nX = ROWS * DMODEL;
  sa_convert_bf16 <<<(nX + 255) / 256, 256, 0, stream>>>(x, xb, nX);
  sa_transpose_bf16<<<(DMODEL * QKV_N + 255) / 256, 256, 0, stream>>>(Wqkv, WqkvT, DMODEL, QKV_N);
  sa_transpose_bf16<<<(DMODEL * DMODEL + 255) / 256, 256, 0, stream>>>(Wout, WoutT, DMODEL, DMODEL);

  // (ROWS/16) * (QKV_N/64) = 512*24 = 12288 waves ; 4 waves/block
  sa_qkv_gemm  <<<12288 / 4, 128, 0, stream>>>(xb, WqkvT, Qb, Kb, Vtb);
  // BH * SEQ/16 = 16*256 = 4096 waves
  sa_flash_attn<<<4096 / 4, 128, 0, stream>>>(Qb, Kb, Vtb, ctx);
  // (ROWS/16) * (DMODEL/64) = 512*8 = 4096 waves
  sa_out_gemm  <<<4096 / 4, 128, 0, stream>>>(ctx, WoutT, bout, x, yres);
  // one wave per row: 8192 waves, 8 waves/block
  sa_layernorm <<<ROWS / 8, 256, 0, stream>>>(yres, gamma, beta, out);
}
